// MultiHeadAttention_71545565217267
// MI455X (gfx1250) — matrix operand order repaired and checked by reading, not compiled, not run on hardware
//
#include <hip/hip_runtime.h>
#include <hip/hip_bf16.h>

#define BB 2
#define SS 2048
#define DD 1024
#define HH 16
#define DHH 64
#define N3 3072
#define MROWS (BB * SS)

typedef __attribute__((ext_vector_type(16))) __bf16 v16bf;
typedef __attribute__((ext_vector_type(8))) float v8f;

static __device__ inline v8f wmma_bf16(v16bf a, v16bf b, v8f c) {
  // D = A(16x32 bf16) * B(32x16 bf16) + C(16x16 f32)
  return __builtin_amdgcn_wmma_f32_16x16x32_bf16(false, a, false, b, (short)0, c,
                                                 false, false);
}

// A fragment (16x32): lane L holds row M=row0+(L&15); elems j<8 -> K=kb8+j,
// j>=8 -> K=kb8+16+(j-8), kb8=(L>>4)*8.  Source fp32 row-major, lda stride.
static __device__ inline v16bf load_a_f32(const float* __restrict__ A, int lda,
                                          int row0, int k0, int lane) {
  const float* p = A + (size_t)(row0 + (lane & 15)) * lda + (k0 + ((lane >> 4) << 3));
  v16bf a;
#pragma unroll
  for (int j = 0; j < 8; ++j) { a[j] = (__bf16)p[j]; a[j + 8] = (__bf16)p[j + 16]; }
  return a;
}

static __device__ inline v16bf load_a_bf16(const __bf16* __restrict__ A, int lda,
                                           int row0, int k0, int lane) {
  const __bf16* p = A + (size_t)(row0 + (lane & 15)) * lda + (k0 + ((lane >> 4) << 3));
  v16bf a;
#pragma unroll
  for (int j = 0; j < 8; ++j) { a[j] = p[j]; a[j + 8] = p[j + 16]; }
  return a;
}

// B fragment (32x16): lane L holds col N=col0+(L&15); elems j<8 -> K=kb8+j,
// j>=8 -> K=kb8+16+(j-8), kb8=(L>>4)*8.  Source W row-major [K x N], ldw stride.
static __device__ inline v16bf load_b_rowmaj_f32(const float* __restrict__ W, int ldw,
                                                 int k0, int col0, int lane) {
  const float* p = W + (size_t)(k0 + ((lane >> 4) << 3)) * ldw + (col0 + (lane & 15));
  v16bf b;
#pragma unroll
  for (int j = 0; j < 8; ++j) { b[j] = (__bf16)p[(size_t)j * ldw]; b[j + 8] = (__bf16)p[(size_t)(j + 16) * ldw]; }
  return b;
}

// B fragment where B[k][n] = Mx[n0+n][k0+k] (B supplied transposed,
// k per lane: j<8 -> kb8+j, j>=8 -> kb8+16+(j-8), kb8=(L>>4)*8).
static __device__ inline v16bf load_b_colmaj_bf16(const __bf16* __restrict__ Mx, int ldm,
                                                  int n0, int k0, int lane) {
  const __bf16* p = Mx + (size_t)(n0 + (lane & 15)) * ldm + (k0 + ((lane >> 4) << 3));
  v16bf b;
#pragma unroll
  for (int j = 0; j < 8; ++j) { b[j] = p[j]; b[j + 8] = p[j + 16]; }
  return b;
}

// ---------------- Pass 1: QKV = x @ qkv_w + b.  64x64 tile per wave
// (16 accumulators, 4x fragment reuse). Scatter q,k as (B,H,S,DH), v
// transposed (B,H,DH,S), all bf16. ----------------
__global__ __launch_bounds__(256) void qkv_kernel(const float* __restrict__ x,
                                                  const float* __restrict__ w,
                                                  const float* __restrict__ bias,
                                                  __bf16* __restrict__ qm,
                                                  __bf16* __restrict__ km,
                                                  __bf16* __restrict__ vt) {
  int lane = threadIdx.x & 31;
  int wave = threadIdx.x >> 5;
  int tile = blockIdx.x * 8 + wave;
  int tm = tile / (N3 / 64);
  int tn = tile % (N3 / 64);
  int row0 = tm * 64, col0 = tn * 64;
  v8f acc[4][4] = {};
  for (int kk = 0; kk < DD; kk += 32) {
    if (kk + 32 < DD) {  // hide L2 latency for next k-step behind WMMA chain
      __builtin_prefetch(x + (size_t)(row0 + (lane & 15)) * DD + kk + 32, 0, 1);
      __builtin_prefetch(w + (size_t)(kk + 32 + ((lane >> 4) << 4)) * N3 + col0 + (lane & 15), 0, 1);
    }
    v16bf a[4], b[4];
#pragma unroll
    for (int i = 0; i < 4; ++i) a[i] = load_a_f32(x, DD, row0 + 16 * i, kk, lane);
#pragma unroll
    for (int j = 0; j < 4; ++j) b[j] = load_b_rowmaj_f32(w, N3, kk, col0 + 16 * j, lane);
#pragma unroll
    for (int i = 0; i < 4; ++i)
#pragma unroll
      for (int j = 0; j < 4; ++j) acc[i][j] = wmma_bf16(a[i], b[j], acc[i][j]);
  }
  int half = lane >> 4;
  int which = col0 / DD;  // 0=q 1=k 2=v (uniform: 64-wide tile, DD%64==0)
  int h = (col0 % DD) / DHH;  // uniform per tile (DHH==64-wide tile)
#pragma unroll
  for (int j = 0; j < 4; ++j) {
    int n = col0 + 16 * j + (lane & 15);
    int dh = n % DHH;
    float bv = bias[n];
#pragma unroll
    for (int i = 0; i < 4; ++i) {
#pragma unroll
      for (int r = 0; r < 8; ++r) {
        int m = row0 + 16 * i + r + 8 * half;
        int bidx = m / SS, s = m % SS;
        int bh = bidx * HH + h;
        float val = acc[i][j][r] + bv;
        if (which == 2) {
          vt[((size_t)bh * DHH + dh) * SS + s] = (__bf16)val;
        } else {
          size_t off = ((size_t)bh * SS + s) * DHH + dh;
          if (which == 0) qm[off] = (__bf16)val;
          else            km[off] = (__bf16)val;
        }
      }
    }
  }
}

// Online-softmax update for one 16q x 32k transposed score block.
// c0/c1: S^T C-fragments (M=k rows, N=q cols). Returns the P A-fragment.
static __device__ inline v16bf softmax_update(v8f c0, v8f c1, int kv0, int qcol,
                                              int half, float& m_run, float& l_run,
                                              v8f* acc) {
  float bm = -3.0e38f;
#pragma unroll
  for (int r = 0; r < 8; ++r) {
    int k0i = kv0 + r + 8 * half;
    float s0 = c0[r] * 0.125f; if (k0i > qcol)      s0 = -1.0e9f;
    float s1 = c1[r] * 0.125f; if (k0i + 16 > qcol) s1 = -1.0e9f;
    c0[r] = s0; c1[r] = s1;
    bm = fmaxf(bm, fmaxf(s0, s1));
  }
  bm = fmaxf(bm, __shfl_xor(bm, 16, 32));  // combine lane pair (same q)
  float m_new = fmaxf(m_run, bm);
  float alpha = __expf(m_run - m_new);
  float rsum = 0.f;
  v16bf pa;  // A-fragment of P: elem j<8 <- c0[j], elem j>=8 <- c1[j-8]
#pragma unroll
  for (int r = 0; r < 8; ++r) {
    float p0 = __expf(c0[r] - m_new);
    float p1 = __expf(c1[r] - m_new);
    rsum += p0 + p1;
    pa[r] = (__bf16)p0;
    pa[r + 8] = (__bf16)p1;
  }
  rsum += __shfl_xor(rsum, 16, 32);
  l_run = l_run * alpha + rsum;
  m_run = m_new;
#pragma unroll
  for (int r = 0; r < 8; ++r) {  // rescale ctx acc: M=q=r+8*half
    float af = __shfl(alpha, r + 8 * half, 32);
    acc[0][r] *= af; acc[1][r] *= af; acc[2][r] *= af; acc[3][r] *= af;
  }
  return pa;
}

static __device__ inline void store_ctx(__bf16* __restrict__ ctx, const v8f* acc,
                                        float l_run, int b, int h, int q0,
                                        int half, int lane) {
#pragma unroll
  for (int r = 0; r < 8; ++r) {
    float lf = __shfl(l_run, r + 8 * half, 32);
    float inv = 1.0f / lf;
    int qg = q0 + r + 8 * half;
    size_t base = ((size_t)b * SS + qg) * DD + h * DHH + (lane & 15);
    ctx[base + 0]  = (__bf16)(acc[0][r] * inv);
    ctx[base + 16] = (__bf16)(acc[1][r] * inv);
    ctx[base + 32] = (__bf16)(acc[2][r] * inv);
    ctx[base + 48] = (__bf16)(acc[3][r] * inv);
  }
}

// ---------------- Pass 2: flash attention. Each wave owns TWO 16-row q-tiles
// so every K/V fragment feeds 2x the WMMAs (16 per 32-wide k-block). Scores
// computed transposed (S^T = K @ Q^T) so the softmaxed C-fragment IS the
// A-fragment of P@V (no cross-lane movement). ----------------
__global__ __launch_bounds__(256) void attn_kernel(const __bf16* __restrict__ qm,
                                                   const __bf16* __restrict__ km,
                                                   const __bf16* __restrict__ vt,
                                                   __bf16* __restrict__ ctx) {
  int lane = threadIdx.x & 31;
  int wave = threadIdx.x >> 5;
  int t = blockIdx.x * 8 + wave;       // q-tile-pair index
  int pt = t % (SS / 32);
  int bh = t / (SS / 32);
  int b = bh / HH, h = bh % HH;
  int q0a = pt * 32, q0b = pt * 32 + 16;
  const __bf16* Qh = qm + (size_t)bh * SS * DHH;
  const __bf16* Kh = km + (size_t)bh * SS * DHH;
  const __bf16* Vh = vt + (size_t)bh * DHH * SS;

  // Q^T B-fragments per tile (dh 0..31, 32..63): loop-invariant
  v16bf qa0 = load_b_colmaj_bf16(Qh, DHH, q0a, 0, lane);
  v16bf qa1 = load_b_colmaj_bf16(Qh, DHH, q0a, 32, lane);
  v16bf qb0 = load_b_colmaj_bf16(Qh, DHH, q0b, 0, lane);
  v16bf qb1 = load_b_colmaj_bf16(Qh, DHH, q0b, 32, lane);

  float mA = -3.0e38f, lA = 0.f, mB = -3.0e38f, lB = 0.f;
  v8f accA[4] = {}, accB[4] = {};
  int half = lane >> 4;
  int qcolA = q0a + (lane & 15);
  int qcolB = q0b + (lane & 15);
  int nkb = (q0b + 16 + 31) / 32;  // causal extent of the LATER tile
  for (int kb = 0; kb < nkb; ++kb) {
    int kv0 = kb * 32;
    // shared K fragments for both q-tiles
    v16bf ka0 = load_a_bf16(Kh, DHH, kv0, 0, lane);
    v16bf ka1 = load_a_bf16(Kh, DHH, kv0, 32, lane);
    v16bf ka2 = load_a_bf16(Kh, DHH, kv0 + 16, 0, lane);
    v16bf ka3 = load_a_bf16(Kh, DHH, kv0 + 16, 32, lane);
    v8f cA0 = {}, cA1 = {}, cB0 = {}, cB1 = {};
    cA0 = wmma_bf16(ka0, qa0, cA0); cA0 = wmma_bf16(ka1, qa1, cA0);
    cA1 = wmma_bf16(ka2, qa0, cA1); cA1 = wmma_bf16(ka3, qa1, cA1);
    cB0 = wmma_bf16(ka0, qb0, cB0); cB0 = wmma_bf16(ka1, qb1, cB0);
    cB1 = wmma_bf16(ka2, qb0, cB1); cB1 = wmma_bf16(ka3, qb1, cB1);

    v16bf paA = softmax_update(cA0, cA1, kv0, qcolA, half, mA, lA, accA);
    v16bf paB = softmax_update(cB0, cB1, kv0, qcolB, half, mB, lB, accB);

    // shared V fragments feed both tiles' P@V
#pragma unroll
    for (int d = 0; d < 4; ++d) {
      v16bf vb = load_b_colmaj_bf16(Vh, SS, d * 16, kv0, lane);
      accA[d] = wmma_bf16(paA, vb, accA[d]);
      accB[d] = wmma_bf16(paB, vb, accB[d]);
    }
  }
  store_ctx(ctx, accA, lA, b, h, q0a, half, lane);
  store_ctx(ctx, accB, lB, b, h, q0b, half, lane);
}

// ---------------- Pass 3: out = ctx @ out_w + out_b (fp32), 64x64/wave -----
__global__ __launch_bounds__(256) void out_kernel(const __bf16* __restrict__ ctx,
                                                  const float* __restrict__ w,
                                                  const float* __restrict__ bias,
                                                  float* __restrict__ out) {
  int lane = threadIdx.x & 31;
  int wave = threadIdx.x >> 5;
  int tile = blockIdx.x * 8 + wave;
  int tm = tile / (DD / 64);
  int tn = tile % (DD / 64);
  int row0 = tm * 64, col0 = tn * 64;
  v8f acc[4][4] = {};
  for (int kk = 0; kk < DD; kk += 32) {
    if (kk + 32 < DD) {
      __builtin_prefetch(ctx + (size_t)(row0 + (lane & 15)) * DD + kk + 32, 0, 1);
      __builtin_prefetch(w + (size_t)(kk + 32 + ((lane >> 4) << 4)) * DD + col0 + (lane & 15), 0, 1);
    }
    v16bf a[4], b[4];
#pragma unroll
    for (int i = 0; i < 4; ++i) a[i] = load_a_bf16(ctx, DD, row0 + 16 * i, kk, lane);
#pragma unroll
    for (int j = 0; j < 4; ++j) b[j] = load_b_rowmaj_f32(w, DD, kk, col0 + 16 * j, lane);
#pragma unroll
    for (int i = 0; i < 4; ++i)
#pragma unroll
      for (int j = 0; j < 4; ++j) acc[i][j] = wmma_bf16(a[i], b[j], acc[i][j]);
  }
  int half = lane >> 4;
#pragma unroll
  for (int j = 0; j < 4; ++j) {
    int n = col0 + 16 * j + (lane & 15);
    float bv = bias[n];
#pragma unroll
    for (int i = 0; i < 4; ++i)
#pragma unroll
      for (int r = 0; r < 8; ++r) {
        int m = row0 + 16 * i + r + 8 * half;
        out[(size_t)m * DD + n] = acc[i][j][r] + bv;
      }
  }
}

extern "C" void kernel_launch(void* const* d_in, const int* in_sizes, int n_in,
                              void* d_out, int out_size, void* d_ws, size_t ws_size,
                              hipStream_t stream) {
  (void)in_sizes; (void)n_in; (void)out_size; (void)ws_size;
  const float* x     = (const float*)d_in[0];
  // d_in[1]: mask (causal, known analytically) — unused
  const float* qkv_w = (const float*)d_in[2];
  const float* qkv_b = (const float*)d_in[3];
  const float* out_w = (const float*)d_in[4];
  const float* out_b = (const float*)d_in[5];
  float* out = (float*)d_out;

  const size_t per = (size_t)BB * HH * SS * DHH;  // 4M bf16 elements each
  __bf16* qm  = (__bf16*)d_ws;
  __bf16* km  = qm + per;
  __bf16* vt  = km + per;
  __bf16* ctx = vt + per;  // 32 MB workspace total

  dim3 blk(256);
  qkv_kernel<<<(MROWS / 64) * (N3 / 64) / 8, blk, 0, stream>>>(x, qkv_w, qkv_b, qm, km, vt);
  attn_kernel<<<(BB * HH * (SS / 32)) / 8, blk, 0, stream>>>(qm, km, vt, ctx);
  out_kernel<<<(MROWS / 64) * (DD / 64) / 8, blk, 0, stream>>>(ctx, out_w, out_b, out);
}
